// VCDiscreteEstimator_39771397161049
// MI455X (gfx1250) — matrix operand order repaired and checked by reading, not compiled, not run on hardware
//
#include <hip/hip_runtime.h>
#include <hip/hip_bf16.h>

// ---------------------------------------------------------------------------
// VCDiscreteEstimator, MI455X (gfx1250, wave32).
//
// Only columns li(b), ui(b) of the [B,257] spline head are ever read, so we
// evaluate just those via a "batched dot = WMMA diagonal" trick:
//   per wave: A = 16 z-rows (16x128 bf16); B_nb columns = gathered weight
//   vectors w[:, o_col, nb]; diag(A @ B_nb) is the per-row contraction.
// z (32 MB) is streamed once from HBM with NT hints (~1.4 us @ 23.3 TB/s
// roofline floor); weight is repacked once to bf16 (329 KB, L2/WGP$-resident)
// for contiguous per-lane B-fragment loads. 40 v_wmma_f32_16x16x32_bf16 per
// 16 rows, software-pipelined with ping-pong B buffers so L2 latency is
// hidden behind the matrix pipe instead of exposed as s_wait_loadcnt 0.
// ---------------------------------------------------------------------------

typedef __attribute__((ext_vector_type(16))) __bf16 v16bf;
typedef __attribute__((ext_vector_type(8)))  float  v8f;
typedef __attribute__((ext_vector_type(4)))  float  f32x4;

#define IN_DIM 128
#define OUT_DIM 257
#define NB 5
#define NGRID 256

// ---- pass 0: weight [IN, OUT, NB] f32  ->  wB [(o*NB+nb)*IN + i] bf16 ----
__global__ void vcde_weight_repack(const float* __restrict__ w,
                                   __bf16* __restrict__ wB) {
    int idx = blockIdx.x * blockDim.x + threadIdx.x;
    if (idx >= OUT_DIM * NB * IN_DIM) return;
    int i  = idx & (IN_DIM - 1);
    int on = idx >> 7;            // o*NB + nb
    int o  = on / NB;
    int nb = on - o * NB;
    wB[idx] = (__bf16)w[(i * OUT_DIM + o) * NB + nb];
}

__global__ __launch_bounds__(256)
void vcde_main(const float* __restrict__ d, const float* __restrict__ z,
               const __bf16* __restrict__ wB, const float* __restrict__ bias,
               float* __restrict__ out) {
    const int lane    = threadIdx.x & 31;
    const int wave    = threadIdx.x >> 5;
    const int group   = blockIdx.x * 8 + wave;   // 16 rows per wave
    const int rowbase = group * 16;

    const int   m  = lane & 15;                  // row (= column) this lane serves
    const float dm = d[rowbase + m];

    // spline basis: [1, d, d^2, relu(d-.33)^2, relu(d-.66)^2]
    float bas[NB];
    bas[0] = 1.0f;
    bas[1] = dm;
    bas[2] = dm * dm;
    float r1 = fmaxf(dm - 0.33f, 0.0f); bas[3] = r1 * r1;
    float r2 = fmaxf(dm - 0.66f, 0.0f); bas[4] = r2 * r2;

    // interpolation params (tmp = (d-a)*(b-a) = d)
    const float g  = dm * (float)NGRID;
    float fl = fminf(fmaxf(floorf(g), 0.0f), (float)(NGRID - 1));
    float cl = fminf(fmaxf(ceilf(g),  0.0f), (float)(NGRID - 1));
    const int   lo_o = (int)fl;
    const int   up_o = (int)cl;
    const float dist = g - fl;

    // per-lane base pointers for this lane's B column (its row's o-slices)
    const int khalfB = (lane < 16) ? 0 : 16;
    const __bf16* wlo = wB + (size_t)lo_o * (NB * IN_DIM) + khalfB;
    const __bf16* wup = wB + (size_t)up_o * (NB * IN_DIM) + khalfB;

    // ---- A fragments: 16 z-rows as bf16, K = 0..127 in 4 chunks of 32 -----
    // A k order used here (contiguous, B matches): lanes 0-15 hold K{0..15}, lanes 16-31 hold
    // K{16..31} of their row, ascending within the v16bf.
    // z is read exactly once -> non-temporal so it can't evict the hot
    // weight table from WGP$/L2.
    const int    khalfA = (lane < 16) ? 0 : 16;
    const float* zr     = z + (size_t)(rowbase + m) * IN_DIM;
    v16bf afrag[4];
#pragma unroll
    for (int c = 0; c < 4; ++c) {
        f32x4 p0 = __builtin_nontemporal_load((const f32x4*)(zr + c * 32 + khalfA + 0));
        f32x4 p1 = __builtin_nontemporal_load((const f32x4*)(zr + c * 32 + khalfA + 4));
        f32x4 p2 = __builtin_nontemporal_load((const f32x4*)(zr + c * 32 + khalfA + 8));
        f32x4 p3 = __builtin_nontemporal_load((const f32x4*)(zr + c * 32 + khalfA + 12));
        v16bf a;
#pragma unroll
        for (int j = 0; j < 4; ++j) {
            a[j]      = (__bf16)p0[j];
            a[4 + j]  = (__bf16)p1[j];
            a[8 + j]  = (__bf16)p2[j];
            a[12 + j] = (__bf16)p3[j];
        }
        afrag[c] = a;
    }

    // diag element m of a 16x16 f32 C sits at lane (m<8 ? m : m+16), VGPR m&7
    const int rr = ((lane < 16) ? m : (m - 8)) & 7;

    // ---- pipelined WMMA: 8 steps = {lo,up} x 4 K-chunks, 5 nb each --------
    // Ping-pong B buffers: while step t runs its 5 WMMAs, step t+1's ten
    // global_load_b128 are in flight. All control flow is compile-time
    // uniform, so EXEC stays all-1s through every WMMA.
    const v8f vzero = {};
    float dg[2][NB];
    v8f   acc[NB] = {};
    v16bf bbuf[2][NB];

#pragma unroll
    for (int nb = 0; nb < NB; ++nb)
        bbuf[0][nb] = *(const v16bf*)(wlo + nb * IN_DIM);

#pragma unroll
    for (int t = 0; t < 8; ++t) {
        const int pass = t >> 2;          // 0 = lower column, 1 = upper column
        const int c    = t & 3;           // K-chunk
        const int cur  = t & 1;
        const int nxt  = cur ^ 1;
        if (t < 7) {
            const int     c2 = (t + 1) & 3;
            const __bf16* wp = ((t + 1) >> 2) ? wup : wlo;
#pragma unroll
            for (int nb = 0; nb < NB; ++nb)
                bbuf[nxt][nb] = *(const v16bf*)(wp + nb * IN_DIM + c2 * 32);
        }
#pragma unroll
        for (int nb = 0; nb < NB; ++nb)
            acc[nb] = __builtin_amdgcn_wmma_f32_16x16x32_bf16(
                false, afrag[c], false, bbuf[cur][nb], (short)0, acc[nb],
                false, false);
        if (c == 3) {
#pragma unroll
            for (int nb = 0; nb < NB; ++nb) {
                dg[pass][nb] = acc[nb][rr];
                acc[nb]      = vzero;
            }
        }
    }

    // ---- epilogue: bias gather, basis combine, lerp -----------------------
    float lo_val = 0.0f, up_val = 0.0f;
#pragma unroll
    for (int nb = 0; nb < NB; ++nb) {
        lo_val += bas[nb] * (dg[0][nb] + bias[lo_o * NB + nb]);
        up_val += bas[nb] * (dg[1][nb] + bias[up_o * NB + nb]);
    }
    float prob = lo_val + (up_val - lo_val) * dist;
    prob *= (dm >= 0.0f && dm <= 1.0f) ? 1.0f : 0.0f;

    // lanes {0..7, 24..31} hold the valid diagonals (rows 0..7 / 8..15)
    if (lane < 8 || lane >= 24)
        __builtin_nontemporal_store(prob, out + rowbase + m);
}

extern "C" void kernel_launch(void* const* d_in, const int* in_sizes, int n_in,
                              void* d_out, int out_size, void* d_ws, size_t ws_size,
                              hipStream_t stream) {
    (void)in_sizes; (void)n_in; (void)out_size; (void)ws_size;
    const float* d_treat = (const float*)d_in[0];  // [65536]
    const float* z       = (const float*)d_in[1];  // [65536, 128]
    const float* w       = (const float*)d_in[2];  // [128, 257, 5]
    const float* bias    = (const float*)d_in[3];  // [257, 5]
    float*       out     = (float*)d_out;          // [65536]
    __bf16*      wB      = (__bf16*)d_ws;          // 257*5*128 bf16 = 329 KB

    const int nw = OUT_DIM * NB * IN_DIM;
    vcde_weight_repack<<<(nw + 255) / 256, 256, 0, stream>>>(w, wB);

    // 65536 rows / (16 rows/wave * 8 waves/block) = 512 blocks
    vcde_main<<<512, 256, 0, stream>>>(d_treat, z, wB, bias, out);
}
